// SonataTTSForExport_43576738185944
// MI455X (gfx1250) — matrix operand order repaired and checked by reading, not compiled, not run on hardware
//
#include <hip/hip_runtime.h>

// ---------------- problem dims ----------------
static constexpr int L_  = 12;
static constexpr int D_  = 512;
static constexpr int H_  = 8;
static constexpr int HD_ = 64;
static constexpr int F_  = 2048;
static constexpr int C_  = 192;
static constexpr int B_  = 4;
static constexpr int T_  = 2048;
static constexpr int BT_ = B_ * T_;   // 8192 rows

typedef __attribute__((ext_vector_type(16))) __bf16 v16bf;
typedef __attribute__((ext_vector_type(8)))  float  v8f;
typedef __attribute__((ext_vector_type(4)))  unsigned int uv4;
typedef __attribute__((ext_vector_type(8)))  int iv8;
typedef __attribute__((ext_vector_type(4)))  int iv4;

union frag_u { uint4 q[2]; v16bf v; };

__device__ __forceinline__ __bf16 f2bf(float f) {
  union { float f; unsigned u; } a; a.f = f;
  unsigned r = a.u + 0x7fffu + ((a.u >> 16) & 1u);
  unsigned short h = (unsigned short)(r >> 16);
  union { unsigned short s; __bf16 b; } o; o.s = h;
  return o.b;
}

// ---------------- TDM: async 2-D bf16 tile load (rows x 32 elems) into LDS -------
// Loads tile_dim1=128 rows x tile_dim0=32 elements (2B each), row stride = K elems,
// with hardware LDS padding of 4 dwords every 16 dwords -> LDS rows of 40 bf16.
__device__ __forceinline__ void tdm_load_tile(unsigned lds_off, const __bf16* g,
                                              unsigned K) {
  unsigned long long ga = (unsigned long long)(size_t)g;
  uv4 g0;
  g0[0] = 1u;                                    // count=1, user descriptor
  g0[1] = lds_off;                               // lds_addr (bytes)
  g0[2] = (unsigned)ga;                          // global_addr[31:0]
  g0[3] = (unsigned)(ga >> 32) | (2u << 30);     // global_addr[56:32] | type=2
  // group1 qwords (256 bits)
  unsigned long long q0 = (1ull << 16)           // data_size = 2 bytes
                        | (1ull << 20)           // pad_enable
                        | (3ull << 22)           // pad_interval: 16 dwords
                        | (3ull << 25)           // pad_amount:   4 dwords
                        | ((unsigned long long)(K & 0xffffu) << 48);   // tensor_dim0 lo
  unsigned long long q1 = (unsigned long long)((K >> 16) & 0xffffu)    // tensor_dim0 hi
                        | (0x100000ull << 16)    // tensor_dim1 = 1M rows (no OOB)
                        | (32ull << 48);         // tile_dim0 = 32 elements
  unsigned long long q2 = 128ull                 // tile_dim1 = 128 rows (tile_dim2=0)
                        | ((unsigned long long)K << 32);  // dim0_stride[31:0]
  unsigned long long q3 = 0ull;                  // dim0_stride[47:32]=0, dim1_stride=0
  iv8 g1;
  g1[0] = (int)q0; g1[1] = (int)(q0 >> 32);
  g1[2] = (int)q1; g1[3] = (int)(q1 >> 32);
  g1[4] = (int)q2; g1[5] = (int)(q2 >> 32);
  g1[6] = (int)q3; g1[7] = (int)(q3 >> 32);
  iv4 z4 = {0, 0, 0, 0};
#if __clang_major__ >= 23
  iv8 z8 = {0, 0, 0, 0, 0, 0, 0, 0};
  __builtin_amdgcn_tensor_load_to_lds(g0, g1, z4, z4, z8, 0);
#else
  __builtin_amdgcn_tensor_load_to_lds(g0, g1, z4, z4, 0);
#endif
}

// ---------------- block-wide (sum, sumsq) reduction over 256 threads ----------------
__device__ __forceinline__ float2 blockReduce2(float a, float b, float2* sm) {
#pragma unroll
  for (int m = 16; m >= 1; m >>= 1) {
    a += __shfl_xor(a, m, 32);
    b += __shfl_xor(b, m, 32);
  }
  int w = threadIdx.x >> 5;
  __syncthreads();
  if ((threadIdx.x & 31) == 0) sm[w] = make_float2(a, b);
  __syncthreads();
  float2 out = make_float2(0.f, 0.f);
#pragma unroll
  for (int i = 0; i < 8; i++) { out.x += sm[i].x; out.y += sm[i].y; }
  return out;
}

// ---------------- embedding gather ----------------
__global__ __launch_bounds__(256) void embed_kernel(const int* __restrict__ ids,
                                                    const float* __restrict__ tbl,
                                                    float* __restrict__ x) {
  int idx = blockIdx.x * 256 + threadIdx.x;      // BT_*D_ total
  int d   = idx & (D_ - 1);
  int row = idx >> 9;
  x[idx] = tbl[(size_t)ids[row] * D_ + d];
}

// ---------------- weight f32 [l][K][N] -> bf16 transposed [l][N][K] ----------------
__global__ __launch_bounds__(256) void cvtT_kernel(const float* __restrict__ src,
                                                   __bf16* __restrict__ dst,
                                                   int K, int N) {
  int idx = blockIdx.x * 256 + threadIdx.x;      // covers L*K*N
  int kn  = K * N;
  int l   = idx / kn;
  int r   = idx - l * kn;
  int kk  = r / N;
  int nn  = r - kk * N;
  dst[(size_t)l * kn + (size_t)nn * K + kk] = f2bf(src[idx]);
}

// ---------------- plain f32 -> bf16 ----------------
__global__ __launch_bounds__(256) void cvt_bf16_kernel(const float* __restrict__ src,
                                                       __bf16* __restrict__ dst, int n) {
  int idx = blockIdx.x * 256 + threadIdx.x;
  if (idx < n) dst[idx] = f2bf(src[idx]);
}

// ---------------- LayerNorm (affine) -> bf16; one block per row ----------------
__global__ __launch_bounds__(256) void ln_bf16_kernel(const float* __restrict__ x,
                                                      const float* __restrict__ g,
                                                      const float* __restrict__ bta,
                                                      __bf16* __restrict__ out) {
  __shared__ float2 sm[8];
  size_t row = blockIdx.x;
  int t0 = threadIdx.x * 2;
  const float* xr = x + row * D_;
  float a0 = xr[t0], a1 = xr[t0 + 1];
  float2 r = blockReduce2(a0 + a1, a0 * a0 + a1 * a1, sm);
  float mu  = r.x * (1.f / D_);
  float var = r.y * (1.f / D_) - mu * mu;
  float inv = rsqrtf(var + 1e-5f);
  out[row * D_ + t0]     = f2bf((a0 - mu) * inv * g[t0]     + bta[t0]);
  out[row * D_ + t0 + 1] = f2bf((a1 - mu) * inv * g[t0 + 1] + bta[t0 + 1]);
}

// ---------------- fused double AdaIN: x = LN(LN(x)*(1+s)+b)*(1+es)+eb ----------------
__global__ __launch_bounds__(256) void adain_kernel(float* __restrict__ x,
                                                    const float* __restrict__ cond) {
  __shared__ float2 sm[8];
  size_t row = blockIdx.x;
  int b = (int)(row >> 11);                       // T_ = 2048
  int t0 = threadIdx.x * 2;
  const float* ss = cond + (size_t)(0 * B_ + b) * D_;
  const float* sb = cond + (size_t)(1 * B_ + b) * D_;
  const float* es = cond + (size_t)(2 * B_ + b) * D_;
  const float* eb = cond + (size_t)(3 * B_ + b) * D_;
  float* xr = x + row * D_;
  float a0 = xr[t0], a1 = xr[t0 + 1];
  float2 r = blockReduce2(a0 + a1, a0 * a0 + a1 * a1, sm);
  float mu = r.x * (1.f / D_);
  float inv = rsqrtf(r.y * (1.f / D_) - mu * mu + 1e-5f);
  a0 = (a0 - mu) * inv * (1.f + ss[t0])     + sb[t0];
  a1 = (a1 - mu) * inv * (1.f + ss[t0 + 1]) + sb[t0 + 1];
  r = blockReduce2(a0 + a1, a0 * a0 + a1 * a1, sm);
  mu = r.x * (1.f / D_);
  inv = rsqrtf(r.y * (1.f / D_) - mu * mu + 1e-5f);
  xr[t0]     = (a0 - mu) * inv * (1.f + es[t0])     + eb[t0];
  xr[t0 + 1] = (a1 - mu) * inv * (1.f + es[t0 + 1]) + eb[t0 + 1];
}

// ---------------- AdaIN condition vectors for all layers ----------------
__global__ __launch_bounds__(256) void cond_kernel(const float* __restrict__ spk,
                                                   const int* __restrict__ emo_id,
                                                   const float* __restrict__ emo_tbl,
                                                   const float* __restrict__ sW0, const float* __restrict__ sB0,
                                                   const float* __restrict__ sW1, const float* __restrict__ sB1,
                                                   const float* __restrict__ eW0, const float* __restrict__ eB0,
                                                   const float* __restrict__ eW1, const float* __restrict__ eB1,
                                                   float* __restrict__ cond) {
  int idx = blockIdx.x * 256 + threadIdx.x;      // L_*4*B_*D_
  int d = idx & (D_ - 1);
  int r = idx >> 9;
  int b = r & 3;  r >>= 2;
  int type = r & 3; r >>= 2;
  int l = r;
  const float* vec = (type < 2) ? (spk + (size_t)b * C_)
                                : (emo_tbl + (size_t)emo_id[b] * C_);
  const float* W; const float* Bv;
  if      (type == 0) { W = sW0 + (size_t)l * C_ * D_; Bv = sB0 + (size_t)l * D_; }
  else if (type == 1) { W = sW1 + (size_t)l * C_ * D_; Bv = sB1 + (size_t)l * D_; }
  else if (type == 2) { W = eW0 + (size_t)l * C_ * D_; Bv = eB0 + (size_t)l * D_; }
  else                { W = eW1 + (size_t)l * C_ * D_; Bv = eB1 + (size_t)l * D_; }
  float acc = Bv[d];
  for (int c = 0; c < C_; c++) acc += vec[c] * W[(size_t)c * D_ + d];
  cond[idx] = acc;
}

// ---------------- WMMA bf16 GEMM with TDM double-buffered tile streaming ---------
// C[M,N] = A[M,K] @ Wt[N,K]^T (+ epilogues)
// MODE 0: Cf = acc + bias
// MODE 1: Cf = resid + acc + bias            (residual add, may alias resid)
// MODE 2: Cb = bf16(acc + bias)
// MODE 3: Cb = bf16( silu(resid) * (acc+bias) )   (SwiGLU fuse, resid = u)
template <int MODE>
__global__ __launch_bounds__(256) void gemm_bf16_kernel(const __bf16* __restrict__ A,
                                                        const __bf16* __restrict__ W,
                                                        const float* __restrict__ bias,
                                                        const float* __restrict__ resid,
                                                        float* __restrict__ Cf,
                                                        __bf16* __restrict__ Cb,
                                                        int M, int N, int K) {
  __shared__ __bf16 As[2][128][40];   // TDM pads rows 32 -> 40 bf16
  __shared__ __bf16 Bs[2][128][40];
  const int tid  = threadIdx.x;
  const int wave = tid >> 5, lane = tid & 31;
  const int lm = lane & 15, lh = lane >> 4;
  const int wm = wave & 3, wn = wave >> 2;
  const int m0 = blockIdx.y * 128, n0 = blockIdx.x * 128;

  const v8f vzero = {};
  v8f acc[2][4];
#pragma unroll
  for (int i = 0; i < 2; i++)
#pragma unroll
    for (int j = 0; j < 4; j++) acc[i][j] = vzero;

  const __bf16* aT = A + (size_t)m0 * K;
  const __bf16* bT = W + (size_t)n0 * K;
  const unsigned aOff = (unsigned)(size_t)&As[0][0][0];
  const unsigned bOff = (unsigned)(size_t)&Bs[0][0][0];
  constexpr unsigned BUFB = 128u * 40u * 2u;   // bytes per LDS buffer

  if (wave == 0) {                 // producer wave issues the first pair of tiles
    tdm_load_tile(aOff, aT, (unsigned)K);
    tdm_load_tile(bOff, bT, (unsigned)K);
  }

  const int nk = K >> 5;
  for (int kt = 0; kt < nk; kt++) {
    const int buf = kt & 1;
    if (wave == 0) __builtin_amdgcn_s_wait_tensorcnt(0);  // current tiles landed
    __syncthreads();               // tiles visible; everyone done with buf^1
    if (wave == 0 && kt + 1 < nk) {                       // stream next tiles
      tdm_load_tile(aOff + (unsigned)(buf ^ 1) * BUFB, aT + (size_t)(kt + 1) * 32, (unsigned)K);
      tdm_load_tile(bOff + (unsigned)(buf ^ 1) * BUFB, bT + (size_t)(kt + 1) * 32, (unsigned)K);
    }

    frag_u fa[2], fb[4];
#pragma unroll
    for (int tm = 0; tm < 2; tm++) {
      const __bf16* p = &As[buf][wm * 32 + tm * 16 + lm][lh * 8];
      fa[tm].q[0] = *reinterpret_cast<const uint4*>(p);
      fa[tm].q[1] = *reinterpret_cast<const uint4*>(p + 16);
    }
#pragma unroll
    for (int tn = 0; tn < 4; tn++) {
      const __bf16* p = &Bs[buf][wn * 64 + tn * 16 + lm][lh * 8];
      fb[tn].q[0] = *reinterpret_cast<const uint4*>(p);
      fb[tn].q[1] = *reinterpret_cast<const uint4*>(p + 16);
    }
#pragma unroll
    for (int tm = 0; tm < 2; tm++)
#pragma unroll
      for (int tn = 0; tn < 4; tn++)
        acc[tm][tn] = __builtin_amdgcn_wmma_f32_16x16x32_bf16(
            false, fa[tm].v, false, fb[tn].v, (short)0, acc[tm][tn], false, false);
  }

#pragma unroll
  for (int tm = 0; tm < 2; tm++) {
#pragma unroll
    for (int tn = 0; tn < 4; tn++) {
      const int c = n0 + wn * 64 + tn * 16 + lm;
      const float bv = bias[c];
      const int rbase = m0 + wm * 32 + tm * 16 + lh * 8;
#pragma unroll
      for (int j = 0; j < 8; j++) {
        const size_t idx = (size_t)(rbase + j) * N + c;
        float val = acc[tm][tn][j] + bv;
        if (MODE == 0) {
          Cf[idx] = val;
        } else if (MODE == 1) {
          Cf[idx] = resid[idx] + val;
        } else if (MODE == 2) {
          Cb[idx] = f2bf(val);
        } else {
          float u = resid[idx];
          float s = u / (1.f + __expf(-u));
          Cb[idx] = f2bf(s * val);
        }
      }
    }
  }
}

// ---------------- flash attention (bf16 QK^T and P@V via WMMA) -------------------
// grid: (T_/128, H_, B_); 256 threads = 8 waves, each wave owns 16 query rows
__global__ __launch_bounds__(256) void attn_kernel(const __bf16* __restrict__ Q,
                                                   const __bf16* __restrict__ Kk,
                                                   const __bf16* __restrict__ V,
                                                   __bf16* __restrict__ O) {
  __shared__ __bf16 Ks[64][72];
  __shared__ __bf16 Vt[64][72];
  __shared__ __bf16 Ps[8][16][72];

  const int qb = blockIdx.x, h = blockIdx.y, b = blockIdx.z;
  const int tid = threadIdx.x, wave = tid >> 5, lane = tid & 31;
  const int lm = lane & 15, lh = lane >> 4;
  const size_t base = (size_t)b * T_ * D_ + (size_t)h * HD_;
  const int q0 = qb * 128 + wave * 16;

  frag_u qf[2];
#pragma unroll
  for (int ks = 0; ks < 2; ks++) {
    const __bf16* p = Q + base + (size_t)(q0 + lm) * D_ + ks * 32 + lh * 8;
    qf[ks].q[0] = *reinterpret_cast<const uint4*>(p);
    qf[ks].q[1] = *reinterpret_cast<const uint4*>(p + 16);
  }

  const v8f vzero = {};
  v8f oacc[4];
#pragma unroll
  for (int tn = 0; tn < 4; tn++) oacc[tn] = vzero;
  float mrow[8], lrow[8];
#pragma unroll
  for (int j = 0; j < 8; j++) { mrow[j] = -1e30f; lrow[j] = 0.f; }

  const int row = tid >> 2, seg = tid & 3;

  for (int kb = 0; kb < T_ / 64; kb++) {
    {
      const __bf16* ksrc = Kk + base + (size_t)(kb * 64 + row) * D_ + seg * 16;
      uint4 k0v = reinterpret_cast<const uint4*>(ksrc)[0];
      uint4 k1v = reinterpret_cast<const uint4*>(ksrc)[1];
      *reinterpret_cast<uint4*>(&Ks[row][seg * 16])     = k0v;
      *reinterpret_cast<uint4*>(&Ks[row][seg * 16 + 8]) = k1v;
      const __bf16* vsrc = V + base + (size_t)(kb * 64 + row) * D_ + seg * 16;
      union { uint4 u[2]; __bf16 e[16]; } tmp;
      tmp.u[0] = reinterpret_cast<const uint4*>(vsrc)[0];
      tmp.u[1] = reinterpret_cast<const uint4*>(vsrc)[1];
#pragma unroll
      for (int j = 0; j < 16; j++) Vt[seg * 16 + j][row] = tmp.e[j];
    }
    __syncthreads();

    v8f sacc[4];
#pragma unroll
    for (int tn = 0; tn < 4; tn++) sacc[tn] = vzero;
#pragma unroll
    for (int ks = 0; ks < 2; ks++) {
      frag_u kf[4];
#pragma unroll
      for (int tn = 0; tn < 4; tn++) {
        const __bf16* p = &Ks[tn * 16 + lm][ks * 32 + lh * 8];
        kf[tn].q[0] = *reinterpret_cast<const uint4*>(p);
        kf[tn].q[1] = *reinterpret_cast<const uint4*>(p + 16);
      }
#pragma unroll
      for (int tn = 0; tn < 4; tn++)
        sacc[tn] = __builtin_amdgcn_wmma_f32_16x16x32_bf16(
            false, qf[ks].v, false, kf[tn].v, (short)0, sacc[tn], false, false);
    }
#pragma unroll
    for (int tn = 0; tn < 4; tn++)
#pragma unroll
      for (int j = 0; j < 8; j++) sacc[tn][j] *= 0.125f;

    float corr[8];
#pragma unroll
    for (int j = 0; j < 8; j++) {
      float mx = sacc[0][j];
#pragma unroll
      for (int tn = 1; tn < 4; tn++) mx = fmaxf(mx, sacc[tn][j]);
#pragma unroll
      for (int m = 8; m >= 1; m >>= 1) mx = fmaxf(mx, __shfl_xor(mx, m, 32));
      float mnew = fmaxf(mrow[j], mx);
      corr[j] = __expf(mrow[j] - mnew);
      mrow[j] = mnew;
    }
    float rs[8];
#pragma unroll
    for (int j = 0; j < 8; j++) rs[j] = 0.f;
#pragma unroll
    for (int tn = 0; tn < 4; tn++)
#pragma unroll
      for (int j = 0; j < 8; j++) {
        float p = __expf(sacc[tn][j] - mrow[j]);
        sacc[tn][j] = p;
        rs[j] += p;
      }
#pragma unroll
    for (int j = 0; j < 8; j++) {
#pragma unroll
      for (int m = 8; m >= 1; m >>= 1) rs[j] += __shfl_xor(rs[j], m, 32);
      lrow[j] = lrow[j] * corr[j] + rs[j];
    }
#pragma unroll
    for (int tn = 0; tn < 4; tn++)
#pragma unroll
      for (int j = 0; j < 8; j++) oacc[tn][j] *= corr[j];

#pragma unroll
    for (int tn = 0; tn < 4; tn++)
#pragma unroll
      for (int j = 0; j < 8; j++)
        Ps[wave][lh * 8 + j][tn * 16 + lm] = f2bf(sacc[tn][j]);
    __syncthreads();

#pragma unroll
    for (int ks = 0; ks < 2; ks++) {
      frag_u pf;
      const __bf16* pp = &Ps[wave][lm][ks * 32 + lh * 8];
      pf.q[0] = *reinterpret_cast<const uint4*>(pp);
      pf.q[1] = *reinterpret_cast<const uint4*>(pp + 16);
      frag_u vf[4];
#pragma unroll
      for (int tn = 0; tn < 4; tn++) {
        const __bf16* pv = &Vt[tn * 16 + lm][ks * 32 + lh * 8];
        vf[tn].q[0] = *reinterpret_cast<const uint4*>(pv);
        vf[tn].q[1] = *reinterpret_cast<const uint4*>(pv + 16);
      }
#pragma unroll
      for (int tn = 0; tn < 4; tn++)
        oacc[tn] = __builtin_amdgcn_wmma_f32_16x16x32_bf16(
            false, pf.v, false, vf[tn].v, (short)0, oacc[tn], false, false);
    }
    __syncthreads();
  }

#pragma unroll
  for (int tn = 0; tn < 4; tn++)
#pragma unroll
    for (int j = 0; j < 8; j++) {
      float val = oacc[tn][j] / lrow[j];
      O[base + (size_t)(q0 + lh * 8 + j) * D_ + tn * 16 + lm] = f2bf(val);
    }
}

// =================================================================================
extern "C" void kernel_launch(void* const* d_in, const int* in_sizes, int n_in,
                              void* d_out, int out_size, void* d_ws, size_t ws_size,
                              hipStream_t stream) {
  (void)in_sizes; (void)n_in; (void)out_size; (void)ws_size;

  const int*   text_ids   = (const int*)  d_in[0];
  const float* speaker    = (const float*)d_in[1];
  const int*   emo_id     = (const int*)  d_in[2];
  const float* text_embed = (const float*)d_in[3];
  const float* emo_tbl    = (const float*)d_in[4];
  const float* ln_g  = (const float*)d_in[5];
  const float* ln_b  = (const float*)d_in[6];
  const float* Wq = (const float*)d_in[7];  const float* bq = (const float*)d_in[8];
  const float* Wk = (const float*)d_in[9];  const float* bk = (const float*)d_in[10];
  const float* Wv = (const float*)d_in[11]; const float* bv = (const float*)d_in[12];
  const float* Wo = (const float*)d_in[13]; const float* bo = (const float*)d_in[14];
  const float* spk_sW = (const float*)d_in[15]; const float* spk_sb = (const float*)d_in[16];
  const float* spk_bW = (const float*)d_in[17]; const float* spk_bb = (const float*)d_in[18];
  const float* emo_sW = (const float*)d_in[19]; const float* emo_sb = (const float*)d_in[20];
  const float* emo_bW = (const float*)d_in[21]; const float* emo_bb = (const float*)d_in[22];
  const float* ffn_g = (const float*)d_in[23]; const float* ffn_b = (const float*)d_in[24];
  const float* w1 = (const float*)d_in[25]; const float* b1 = (const float*)d_in[26];
  const float* w2 = (const float*)d_in[27]; const float* b2 = (const float*)d_in[28];
  const float* w3 = (const float*)d_in[29]; const float* b3 = (const float*)d_in[30];
  const float* Wout = (const float*)d_in[31]; const float* bout = (const float*)d_in[32];

  char* ws = (char*)d_ws;
  size_t off = 0;
  auto take = [&](size_t bytes) -> char* {
    char* p = ws + off;
    off += (bytes + 255) & ~(size_t)255;
    return p;
  };

  __bf16* tWq = (__bf16*)take((size_t)L_ * D_ * D_ * 2);
  __bf16* tWk = (__bf16*)take((size_t)L_ * D_ * D_ * 2);
  __bf16* tWv = (__bf16*)take((size_t)L_ * D_ * D_ * 2);
  __bf16* tWo = (__bf16*)take((size_t)L_ * D_ * D_ * 2);
  __bf16* tW1 = (__bf16*)take((size_t)L_ * D_ * F_ * 2);
  __bf16* tW2 = (__bf16*)take((size_t)L_ * D_ * F_ * 2);
  __bf16* tW3 = (__bf16*)take((size_t)L_ * F_ * D_ * 2);
  __bf16* tWout = (__bf16*)take((size_t)D_ * 1024 * 2);
  float*  X  = (float*)take((size_t)BT_ * D_ * 4);
  __bf16* Hb = (__bf16*)take((size_t)BT_ * D_ * 2);
  __bf16* Qb = (__bf16*)take((size_t)BT_ * D_ * 2);
  __bf16* Kb = (__bf16*)take((size_t)BT_ * D_ * 2);
  __bf16* Vb = (__bf16*)take((size_t)BT_ * D_ * 2);
  __bf16* Ab = (__bf16*)take((size_t)BT_ * D_ * 2);
  float*  U  = (float*)take((size_t)BT_ * F_ * 4);
  __bf16* G  = (__bf16*)take((size_t)BT_ * F_ * 2);
  float*  cond = (float*)take((size_t)L_ * 4 * B_ * D_ * 4);

  // ---- prep: embed, weight convert/transpose, AdaIN condition vectors ----
  embed_kernel<<<(BT_ * D_) / 256, 256, 0, stream>>>(text_ids, text_embed, X);
  cvtT_kernel<<<(L_ * D_ * D_) / 256, 256, 0, stream>>>(Wq, tWq, D_, D_);
  cvtT_kernel<<<(L_ * D_ * D_) / 256, 256, 0, stream>>>(Wk, tWk, D_, D_);
  cvtT_kernel<<<(L_ * D_ * D_) / 256, 256, 0, stream>>>(Wv, tWv, D_, D_);
  cvtT_kernel<<<(L_ * D_ * D_) / 256, 256, 0, stream>>>(Wo, tWo, D_, D_);
  cvtT_kernel<<<(L_ * D_ * F_) / 256, 256, 0, stream>>>(w1, tW1, D_, F_);
  cvtT_kernel<<<(L_ * D_ * F_) / 256, 256, 0, stream>>>(w2, tW2, D_, F_);
  cvtT_kernel<<<(L_ * F_ * D_) / 256, 256, 0, stream>>>(w3, tW3, F_, D_);
  cvtT_kernel<<<(D_ * 1024) / 256, 256, 0, stream>>>(Wout, tWout, D_, 1024);
  cond_kernel<<<(L_ * 4 * B_ * D_) / 256, 256, 0, stream>>>(
      speaker, emo_id, emo_tbl, spk_sW, spk_sb, spk_bW, spk_bb,
      emo_sW, emo_sb, emo_bW, emo_bb, cond);

  dim3 gD(D_ / 128, BT_ / 128);
  dim3 gF(F_ / 128, BT_ / 128);

  for (int l = 0; l < L_; l++) {
    // attention block
    ln_bf16_kernel<<<BT_, 256, 0, stream>>>(X, ln_g + (size_t)l * D_, ln_b + (size_t)l * D_, Hb);
    gemm_bf16_kernel<2><<<gD, 256, 0, stream>>>(Hb, tWq + (size_t)l * D_ * D_, bq + (size_t)l * D_,
                                                nullptr, nullptr, Qb, BT_, D_, D_);
    gemm_bf16_kernel<2><<<gD, 256, 0, stream>>>(Hb, tWk + (size_t)l * D_ * D_, bk + (size_t)l * D_,
                                                nullptr, nullptr, Kb, BT_, D_, D_);
    gemm_bf16_kernel<2><<<gD, 256, 0, stream>>>(Hb, tWv + (size_t)l * D_ * D_, bv + (size_t)l * D_,
                                                nullptr, nullptr, Vb, BT_, D_, D_);
    attn_kernel<<<dim3(T_ / 128, H_, B_), 256, 0, stream>>>(Qb, Kb, Vb, Ab);
    gemm_bf16_kernel<1><<<gD, 256, 0, stream>>>(Ab, tWo + (size_t)l * D_ * D_, bo + (size_t)l * D_,
                                                X, X, nullptr, BT_, D_, D_);
    // speaker + emotion AdaIN (fused)
    adain_kernel<<<BT_, 256, 0, stream>>>(X, cond + (size_t)l * 4 * B_ * D_);
    // SwiGLU FFN
    ln_bf16_kernel<<<BT_, 256, 0, stream>>>(X, ffn_g + (size_t)l * D_, ffn_b + (size_t)l * D_, Hb);
    gemm_bf16_kernel<0><<<gF, 256, 0, stream>>>(Hb, tW1 + (size_t)l * F_ * D_, b1 + (size_t)l * F_,
                                                nullptr, U, nullptr, BT_, F_, D_);
    gemm_bf16_kernel<3><<<gF, 256, 0, stream>>>(Hb, tW2 + (size_t)l * F_ * D_, b2 + (size_t)l * F_,
                                                U, nullptr, G, BT_, F_, D_);
    gemm_bf16_kernel<1><<<gD, 256, 0, stream>>>(G, tW3 + (size_t)l * D_ * F_, b3 + (size_t)l * D_,
                                                X, X, nullptr, BT_, D_, F_);
  }

  // final projection to logits
  cvt_bf16_kernel<<<(BT_ * D_) / 256, 256, 0, stream>>>(X, Hb, BT_ * D_);
  gemm_bf16_kernel<0><<<dim3(1024 / 128, BT_ / 128), 256, 0, stream>>>(
      Hb, tWout, bout, nullptr, (float*)d_out, nullptr, BT_, 1024, D_);
}
